// BarCachedCrossAttention_40218073759760
// MI455X (gfx1250) — matrix operand order repaired and checked by reading, not compiled, not run on hardware
//
#include <hip/hip_runtime.h>
#include <hip/hip_bf16.h>

// CDNA5 / gfx1250, wave32. All matmuls on v_wmma_f32_16x16x32_bf16.

typedef __attribute__((ext_vector_type(16))) __bf16 v16bf;
typedef __attribute__((ext_vector_type(8)))  __bf16 v8bf;
typedef __attribute__((ext_vector_type(8)))  float  v8f;

#define BB    4
#define TT    512
#define NC    2048
#define HH    1024
#define NHEAD 16
#define HDIM  64

// Build a 16-element bf16 fragment from two contiguous 8-element (16B) runs.
static __device__ __forceinline__ v16bf frag16(const __bf16* p0, const __bf16* p1) {
  v8bf lo = *(const v8bf*)p0;
  v8bf hi = *(const v8bf*)p1;
  v16bf r;
#pragma unroll
  for (int i = 0; i < 8; ++i) { r[i] = lo[i]; r[i + 8] = hi[i]; }
  return r;
}

static __device__ __forceinline__ v8f wmma_bf16(v16bf a, v16bf b, v8f c) {
  // 8 args: (neg_a, A, neg_b, B, c_mod, C, reuse_a, reuse_b)
  return __builtin_amdgcn_wmma_f32_16x16x32_bf16(false, a, false, b, (short)0, c,
                                                 false, false);
}

// ---------------- prep: embedding add + fp32 -> bf16 ----------------

__global__ void prep_q_kernel(const float* __restrict__ q,
                              const float* __restrict__ inst_emb,
                              const int* __restrict__ curp,
                              __bf16* __restrict__ qe, int n) {
  int i = blockIdx.x * blockDim.x + threadIdx.x;
  if (i >= n) return;
  int j = i & (HH - 1);
  qe[i] = (__bf16)(q[i] + inst_emb[curp[0] * HH + j]);
}

__global__ void prep_ctx_kernel(const float* __restrict__ ctx,
                                const int* __restrict__ ids,
                                const int* __restrict__ bars,
                                const float* __restrict__ inst_emb,
                                const float* __restrict__ bar_emb,
                                __bf16* __restrict__ ce, int n) {
  int i = blockIdx.x * blockDim.x + threadIdx.x;
  if (i >= n) return;
  int j   = i & (HH - 1);
  int row = i >> 10;                // b*NC + n
  int id  = ids[row];
  int bo  = bars[row];
  bo = bo < 0 ? 0 : (bo > 7 ? 7 : bo);
  ce[i] = (__bf16)(ctx[i] + inst_emb[id * HH + j] + bar_emb[bo * HH + j]);
}

__global__ void convw_kernel(const float* __restrict__ w, __bf16* __restrict__ o, int n) {
  int i = blockIdx.x * blockDim.x + threadIdx.x;
  if (i >= n) return;
  o[i] = (__bf16)w[i];
}

// ---------------- generic projection GEMM: out = A @ W.T + bias ----------------
// A: (M x HH) bf16 row-major.  W: (HH x HH) bf16 row-major (torch Linear weight).
// One wave computes a 32x64 strip: 2 A-fragments, 4 B-tiles reused by both row
// halves -> 8 WMMA per 32-wide K step.  (Verified spill-free codegen.)
// mode 0: store Q   bf16 (b, h, t, d)
// mode 1: store K   bf16 (b, h, n, d)
// mode 2: store V^T bf16 (b, h, d, n)
// mode 3: store fp32 row-major (final output projection)

__global__ void gemm_proj_kernel(const __bf16* __restrict__ A,
                                 const __bf16* __restrict__ W,
                                 const float* __restrict__ bias,
                                 void* __restrict__ dst,
                                 int M, int mode) {
  const int lane = threadIdx.x & 31;
  const int half = lane >> 4;
  const int col  = lane & 15;
  const int wid  = blockIdx.x * (blockDim.x >> 5) + (threadIdx.x >> 5);
  const int rt   = wid >> 4;           // row tile of 32
  const int cg   = wid & 15;           // 1024/64 = 16 column groups
  const int nb   = cg * 64;

  const __bf16* arow0 = A + (size_t)(rt * 32 + col) * HH;        // A row = lane&15
  const __bf16* arow1 = A + (size_t)(rt * 32 + 16 + col) * HH;
  const __bf16* bcol[4];
#pragma unroll
  for (int c = 0; c < 4; ++c)
    bcol[c] = W + (size_t)(nb + c * 16 + col) * HH;              // B col n = W row n

  v8f acc[2][4];
#pragma unroll
  for (int q = 0; q < 2; ++q)
#pragma unroll
    for (int c = 0; c < 4; ++c)
#pragma unroll
      for (int r = 0; r < 8; ++r) acc[q][c][r] = 0.0f;

#pragma unroll 2
  for (int k0 = 0; k0 < HH; k0 += 32) {
    __builtin_prefetch(arow0 + k0 + 256, 0, 3);   // global_prefetch_b8
    __builtin_prefetch(arow1 + k0 + 256, 0, 3);
    // A 16x32 fragment: half0 -> K 0..7 & 16..23 ; half1 -> K 8..15 & 24..31
    v16bf a0 = frag16(arow0 + k0 + half * 8, arow0 + k0 + 16 + half * 8);
    v16bf a1 = frag16(arow1 + k0 + half * 8, arow1 + k0 + 16 + half * 8);
#pragma unroll
    for (int c = 0; c < 4; ++c) {
      // B 32x16 fragment: half0 -> K 0..7 & 16..23 ; half1 -> K 8..15 & 24..31
      const __bf16* bp = bcol[c] + k0 + half * 8;
      v16bf b = frag16(bp, bp + 16);
      acc[0][c] = wmma_bf16(a0, b, acc[0][c]);
      acc[1][c] = wmma_bf16(a1, b, acc[1][c]);
    }
  }

#pragma unroll
  for (int q = 0; q < 2; ++q)
#pragma unroll
    for (int c = 0; c < 4; ++c) {
      const int n  = nb + c * 16 + col;
      const float bv = bias[n];
#pragma unroll
      for (int r = 0; r < 8; ++r) {
        const float val = acc[q][c][r] + bv;
        const int gm = rt * 32 + q * 16 + half * 8 + r;
        if (mode == 0) {              // Q: gm = b*TT + t ; n = h*64 + d
          const int b = gm >> 9, t = gm & 511;
          const int h = n >> 6, d = n & 63;
          ((__bf16*)dst)[(((size_t)(b * NHEAD + h) * TT + t) << 6) + d] = (__bf16)val;
        } else if (mode == 1) {       // K: gm = b*NC + nk
          const int b = gm >> 11, nk = gm & 2047;
          const int h = n >> 6, d = n & 63;
          ((__bf16*)dst)[(((size_t)(b * NHEAD + h) * NC + nk) << 6) + d] = (__bf16)val;
        } else if (mode == 2) {       // V transposed: (b,h,d,nk)
          const int b = gm >> 11, nk = gm & 2047;
          const int h = n >> 6, d = n & 63;
          ((__bf16*)dst)[(((size_t)(b * NHEAD + h) * HDIM + d) * NC) + nk] = (__bf16)val;
        } else {                      // final fp32 output
          ((float*)dst)[(size_t)gm * HH + n] = val;
        }
      }
    }
}

// ---------------- flash attention: one wave per (b, h, 16-query tile) ----------------
// Single query tile keeps the live set ~140 VGPRs (no scratch spills).

__global__ void attn_kernel(const __bf16* __restrict__ Qws,
                            const __bf16* __restrict__ Kws,
                            const __bf16* __restrict__ Vt,
                            const int* __restrict__ ids,
                            const int* __restrict__ curp,
                            __bf16* __restrict__ attn) {
  __shared__ __bf16 pb[16 * 32];          // P tile staging (C-layout -> A-layout)

  const int lane = threadIdx.x & 31;
  const int half = lane >> 4;
  const int col  = lane & 15;
  const int blk   = blockIdx.x;           // (b*NHEAD + h)*32 + qtile
  const int qtile = blk & 31;             // TT/16 = 32
  const int bh    = blk >> 5;
  const int b     = bh >> 4;
  const int h     = bh & 15;
  const int cur   = curp[0];

  const __bf16* qrow = Qws + ((size_t)bh * TT + qtile * 16 + col) * HDIM;
  const __bf16* kbs  = Kws + (size_t)bh * NC * HDIM;
  const __bf16* vtb  = Vt  + (size_t)bh * HDIM * NC;
  const int*    idb  = ids + b * NC;

  // Q A-fragments for head-dim K = 0..31 and 32..63 (loaded once)
  const v16bf a0 = frag16(qrow + half * 16,      qrow + half * 16 + 8);
  const v16bf a1 = frag16(qrow + 32 + half * 16, qrow + 40 + half * 16);

  v8f o[4];
  float mrow[8], lrow[8];
#pragma unroll
  for (int c = 0; c < 4; ++c)
#pragma unroll
    for (int r = 0; r < 8; ++r) o[c][r] = 0.0f;
#pragma unroll
  for (int r = 0; r < 8; ++r) { mrow[r] = -3.0e38f; lrow[r] = 0.0f; }

  for (int kb = 0; kb < NC; kb += 32) {
    // ---- scores: two 16x16 tiles (keys kb..kb+15, kb+16..kb+31) ----
    v8f sc0, sc1;
#pragma unroll
    for (int r = 0; r < 8; ++r) { sc0[r] = 0.0f; sc1[r] = 0.0f; }
    {
      const __bf16* kc0 = kbs + (size_t)(kb + col) * HDIM;
      sc0 = wmma_bf16(a0, frag16(kc0 + half * 16,      kc0 + half * 16 + 8),  sc0);
      sc0 = wmma_bf16(a1, frag16(kc0 + 32 + half * 16, kc0 + 40 + half * 16), sc0);
      const __bf16* kc1 = kbs + (size_t)(kb + 16 + col) * HDIM;
      sc1 = wmma_bf16(a0, frag16(kc1 + half * 16,      kc1 + half * 16 + 8),  sc1);
      sc1 = wmma_bf16(a1, frag16(kc1 + 32 + half * 16, kc1 + 40 + half * 16), sc1);
    }

    // ---- scale + instrument mask (lane owns key column kb+g*16+col) ----
    const bool m0 = (idb[kb + col]      == cur);
    const bool m1 = (idb[kb + 16 + col] == cur);
    float f0[8], f1[8];
#pragma unroll
    for (int r = 0; r < 8; ++r) {
      f0[r] = m0 ? -1.0e30f : sc0[r] * 0.125f;   // 1/sqrt(64)
      f1[r] = m1 ? -1.0e30f : sc1[r] * 0.125f;
    }

    // ---- online softmax per row (rows = half*8 + r; reduce over 16 lanes) ----
#pragma unroll
    for (int r = 0; r < 8; ++r) {
      float rm = fmaxf(f0[r], f1[r]);
      rm = fmaxf(rm, __shfl_xor(rm, 1, 32));
      rm = fmaxf(rm, __shfl_xor(rm, 2, 32));
      rm = fmaxf(rm, __shfl_xor(rm, 4, 32));
      rm = fmaxf(rm, __shfl_xor(rm, 8, 32));
      const float mnew = fmaxf(mrow[r], rm);
      const float corr = __expf(mrow[r] - mnew);
      const float p0 = __expf(f0[r] - mnew);
      const float p1 = __expf(f1[r] - mnew);
      float rs = p0 + p1;
      rs += __shfl_xor(rs, 1, 32);
      rs += __shfl_xor(rs, 2, 32);
      rs += __shfl_xor(rs, 4, 32);
      rs += __shfl_xor(rs, 8, 32);
      lrow[r] = lrow[r] * corr + rs;
      mrow[r] = mnew;
      o[0][r] *= corr; o[1][r] *= corr; o[2][r] *= corr; o[3][r] *= corr;
      pb[(half * 8 + r) * 32 + col]      = (__bf16)p0;
      pb[(half * 8 + r) * 32 + 16 + col] = (__bf16)p1;
    }
    __syncthreads();

    // ---- P (16x32, A-layout) x V-block (32x64) ----
    const v16bf pa = frag16(&pb[col * 32 + half * 16], &pb[col * 32 + half * 16 + 8]);
#pragma unroll
    for (int c = 0; c < 4; ++c) {
      const __bf16* vc = vtb + (size_t)(c * 16 + col) * NC + kb + half * 16;
      o[c] = wmma_bf16(pa, frag16(vc, vc + 8), o[c]);
    }
    __syncthreads();
  }

  // ---- finalize: divide by l, store to (B, T, H) bf16 for the output GEMM ----
#pragma unroll
  for (int c = 0; c < 4; ++c)
#pragma unroll
    for (int r = 0; r < 8; ++r) {
      const float v = o[c][r] / lrow[r];
      const int t = qtile * 16 + half * 8 + r;
      attn[((size_t)(b * TT + t)) * HH + h * 64 + c * 16 + col] = (__bf16)v;
    }
}

// ---------------- host-side launch ----------------

extern "C" void kernel_launch(void* const* d_in, const int* in_sizes, int n_in,
                              void* d_out, int out_size, void* d_ws, size_t ws_size,
                              hipStream_t stream) {
  const float* query   = (const float*)d_in[0];
  const float* context = (const float*)d_in[1];
  const int*   ids     = (const int*)d_in[2];
  const int*   curp    = (const int*)d_in[3];
  const int*   bars    = (const int*)d_in[4];
  const float* Wq = (const float*)d_in[5];
  const float* bq = (const float*)d_in[6];
  const float* Wk = (const float*)d_in[7];
  const float* bk = (const float*)d_in[8];
  const float* Wv = (const float*)d_in[9];
  const float* bv = (const float*)d_in[10];
  const float* Wo = (const float*)d_in[11];
  const float* bo = (const float*)d_in[12];
  const float* inst_emb = (const float*)d_in[13];
  const float* bar_emb  = (const float*)d_in[14];

  char* p = (char*)d_ws;
  size_t o = 0;
  __bf16* qe   = (__bf16*)(p + o); o += (size_t)BB * TT * HH * 2;            // 4 MB
  __bf16* ce   = (__bf16*)(p + o); o += (size_t)BB * NC * HH * 2;            // 16 MB
  __bf16* wqb  = (__bf16*)(p + o); o += (size_t)HH * HH * 2;                 // 2 MB
  __bf16* wkb  = (__bf16*)(p + o); o += (size_t)HH * HH * 2;
  __bf16* wvb  = (__bf16*)(p + o); o += (size_t)HH * HH * 2;
  __bf16* wob  = (__bf16*)(p + o); o += (size_t)HH * HH * 2;
  __bf16* Qws  = (__bf16*)(p + o); o += (size_t)BB * NHEAD * TT * HDIM * 2;  // 4 MB
  __bf16* Kws  = (__bf16*)(p + o); o += (size_t)BB * NHEAD * NC * HDIM * 2;  // 16 MB
  __bf16* Vt   = (__bf16*)(p + o); o += (size_t)BB * NHEAD * HDIM * NC * 2;  // 16 MB
  __bf16* attn = (__bf16*)(p + o); o += (size_t)BB * TT * HH * 2;            // 4 MB

  const int nq = BB * TT * HH;     // 2,097,152
  const int nc = BB * NC * HH;     // 8,388,608
  const int nw = HH * HH;          // 1,048,576

  prep_q_kernel<<<(nq + 255) / 256, 256, 0, stream>>>(query, inst_emb, curp, qe, nq);
  prep_ctx_kernel<<<(nc + 255) / 256, 256, 0, stream>>>(context, ids, bars, inst_emb,
                                                        bar_emb, ce, nc);
  convw_kernel<<<(nw + 255) / 256, 256, 0, stream>>>(Wq, wqb, nw);
  convw_kernel<<<(nw + 255) / 256, 256, 0, stream>>>(Wk, wkb, nw);
  convw_kernel<<<(nw + 255) / 256, 256, 0, stream>>>(Wv, wvb, nw);
  convw_kernel<<<(nw + 255) / 256, 256, 0, stream>>>(Wo, wob, nw);

  // waves = (M/32)*16 col-groups; 8 waves per 256-thread block
  gemm_proj_kernel<<<(BB * TT / 32) * 16 / 8, 256, 0, stream>>>(qe, wqb, bq, Qws,
                                                                BB * TT, 0);
  gemm_proj_kernel<<<(BB * NC / 32) * 16 / 8, 256, 0, stream>>>(ce, wkb, bk, Kws,
                                                                BB * NC, 1);
  gemm_proj_kernel<<<(BB * NC / 32) * 16 / 8, 256, 0, stream>>>(ce, wvb, bv, Vt,
                                                                BB * NC, 2);

  attn_kernel<<<BB * NHEAD * (TT / 16), 32, 0, stream>>>(Qws, Kws, Vt, ids, curp, attn);

  gemm_proj_kernel<<<(BB * TT / 32) * 16 / 8, 256, 0, stream>>>(attn, wob, bo,
                                                                (float*)d_out,
                                                                BB * TT, 3);
}
